// StockFormer_69295002353780
// MI455X (gfx1250) — matrix operand order repaired and checked by reading, not compiled, not run on hardware
//
#include <hip/hip_runtime.h>
#include <hip/hip_bf16.h>

// ---------------------------------------------------------------------------
// StockFormer forward on MI455X (gfx1250, wave32).
// All large GEMMs use v_wmma_f32_16x16x32_f16 (f16 in, f32 accumulate) with
// 1Mx4N register blocking per wave (4 WMMAs per A-fragment load).
// B (weight) matrices are pre-swizzled into the WMMA fragment layout so each
// lane's fragment is a single 32B contiguous run (2x b128 loads per WMMA).
// Attention cores (<20% of FLOPs) run in VALU f32 with LDS-staged K/V.
// ---------------------------------------------------------------------------

typedef _Float16 half_t;
typedef __attribute__((ext_vector_type(16))) _Float16 v16h;
typedef __attribute__((ext_vector_type(8)))  _Float16 v8h;
typedef __attribute__((ext_vector_type(8)))  float    v8f;
typedef __attribute__((ext_vector_type(4)))  unsigned int u32x4;
typedef __attribute__((ext_vector_type(8)))  int          i32x8;
typedef __attribute__((ext_vector_type(4)))  int          i32x4;

#define TOKENS 65536   // B*L*S = 4*128*128
#define DM     256
#define FD     362
#define FP     384     // F padded to multiple of 32
#define TT     252
#define LL     128
#define SS     128
#define NHEAD  8
#define HD     32      // head dim

enum { GF_RELU = 1, GF_OUTH = 2, GF_ACC = 4 };

// ---- WMMA fragment loaders (ISA 7.12.2 16-bit layouts, wave32) -------------
// A (16x32, MxK): lane m=l&15, hi=l>>4; halves 0..7 = A[m][8*hi+0..7],
//                 halves 8..15 = A[m][16+8*hi+0..7]  (two contiguous 16B runs)
__device__ __forceinline__ v16h load_a_frag(const half_t* __restrict__ A, int ldA,
                                            int mBase, int k0, int lane) {
  int m  = mBase + (lane & 15);
  int hi = lane >> 4;
  const half_t* p = A + (size_t)m * ldA + k0 + 8 * hi;
  union { v16h v; v8h h[2]; } u;
  u.h[0] = *(const v8h*)(p);
  u.h[1] = *(const v8h*)(p + 16);
  return u.v;
}
// B pre-swizzled: [kBlock][nTile][lane][16 halves] -> one 32B run per lane
__device__ __forceinline__ v16h load_b_frag_swz(const half_t* __restrict__ Bm,
                                                int tilesNB, int nTile, int k0,
                                                int lane) {
  const half_t* p = Bm + ((((size_t)(k0 >> 5) * tilesNB + nTile) << 5) + lane) * 16;
  union { v16h v; v8h h[2]; } u;
  u.h[0] = *(const v8h*)(p);
  u.h[1] = *(const v8h*)(p + 8);
  return u.v;
}

// ---- generic WMMA GEMM: C[M,N] = A[M,K] @ B[K,N] + bias -------------------
// One wave per 16x64 C strip (1M x 4N register blocking); 8 waves per block.
// Requires N % 64 == 0.
__global__ void __launch_bounds__(256)
gemm_wmma_f16(const half_t* __restrict__ A, int ldA,
              const half_t* __restrict__ Bm, int tilesNB, int nbOff,
              const float* __restrict__ bias,
              void* __restrict__ Cout, int ldC,
              int M, int N, int K, int flags) {
  int lane = threadIdx.x & 31;
  int wave = threadIdx.x >> 5;
  int tilesN4 = N >> 6;
  int gw = blockIdx.x * (blockDim.x >> 5) + wave;
  int tileM  = gw / tilesN4;
  int tileN4 = gw - tileM * tilesN4;
  if ((tileM << 4) >= M) return;
  int mBase = tileM << 4;

  v8f acc0 = {0.f, 0.f, 0.f, 0.f, 0.f, 0.f, 0.f, 0.f};
  v8f acc1 = acc0, acc2 = acc0, acc3 = acc0;
  for (int k0 = 0; k0 < K; k0 += 32) {
    if (k0 + 32 < K)  // lowers to global_prefetch_b8 on the A stream
      __builtin_prefetch(A + (size_t)(mBase + (lane & 15)) * ldA + k0 + 32, 0, 1);
    v16h a  = load_a_frag(A, ldA, mBase, k0, lane);
    int nb  = nbOff + (tileN4 << 2);
    v16h b0 = load_b_frag_swz(Bm, tilesNB, nb + 0, k0, lane);
    v16h b1 = load_b_frag_swz(Bm, tilesNB, nb + 1, k0, lane);
    v16h b2 = load_b_frag_swz(Bm, tilesNB, nb + 2, k0, lane);
    v16h b3 = load_b_frag_swz(Bm, tilesNB, nb + 3, k0, lane);
    acc0 = __builtin_amdgcn_wmma_f32_16x16x32_f16(false, a, false, b0, (short)0, acc0, false, false);
    acc1 = __builtin_amdgcn_wmma_f32_16x16x32_f16(false, a, false, b1, (short)0, acc1, false, false);
    acc2 = __builtin_amdgcn_wmma_f32_16x16x32_f16(false, a, false, b2, (short)0, acc2, false, false);
    acc3 = __builtin_amdgcn_wmma_f32_16x16x32_f16(false, a, false, b3, (short)0, acc3, false, false);
  }

  int hi = lane >> 4;
  float*  Cf = (float*)Cout;
  half_t* Ch = (half_t*)Cout;
  v8f accs[4] = {acc0, acc1, acc2, acc3};
  #pragma unroll
  for (int u = 0; u < 4; ++u) {
    int n = (tileN4 << 6) + (u << 4) + (lane & 15);
    float bv = bias ? bias[n] : 0.0f;
    #pragma unroll
    for (int r = 0; r < 8; ++r) {
      int m = mBase + 8 * hi + r;
      float v = accs[u][r] + bv;
      if (flags & GF_ACC)  v += Cf[(size_t)m * ldC + n];
      if (flags & GF_RELU) v = v > 0.f ? v : 0.f;
      if (flags & GF_OUTH) Ch[(size_t)m * ldC + n] = (half_t)v;
      else                 Cf[(size_t)m * ldC + n] = v;
    }
  }
}

// ---- dilated conv1d (k=3, dilation=2, pad=2) as 3-tap accumulated WMMA ----
// token = (b*L + l)*S + s; a 16-row tile has fixed (b,l) since 16 | S.
// 1Mx4N register blocking (N=256 -> 4 strips of 64).
__global__ void __launch_bounds__(256)
conv_gemm_wmma(const half_t* __restrict__ Xh,   // TOKENS x 256 (f16)
               const half_t* __restrict__ Wt,   // 3 x swizzled (256x256) B-mats
               const float* __restrict__ bias,  // conv_b
               half_t* __restrict__ Out) {      // TOKENS x 256 (relu'd, f16)
  int lane = threadIdx.x & 31;
  int wave = threadIdx.x >> 5;
  int gw = blockIdx.x * (blockDim.x >> 5) + wave;
  int tileM  = gw >> 2;           // 4 strips of 64 along N (256/64)
  int tileN4 = gw & 3;
  if ((tileM << 4) >= TOKENS) return;
  int l = ((tileM << 4) / SS) % LL;

  v8f acc0 = {0.f, 0.f, 0.f, 0.f, 0.f, 0.f, 0.f, 0.f};
  v8f acc1 = acc0, acc2 = acc0, acc3 = acc0;
  for (int t = 0; t < 3; ++t) {
    int dl = 2 * (t - 1);
    int ls = l + dl;
    if (ls < 0 || ls >= LL) continue;     // zero padding in l
    int mShift = dl * SS;                 // row shift of the A operand
    const half_t* Bt = Wt + (size_t)t * DM * DM;
    for (int k0 = 0; k0 < DM; k0 += 32) {
      v16h a  = load_a_frag(Xh, DM, (tileM << 4) + mShift, k0, lane);
      int nb  = tileN4 << 2;
      v16h b0 = load_b_frag_swz(Bt, DM >> 4, nb + 0, k0, lane);
      v16h b1 = load_b_frag_swz(Bt, DM >> 4, nb + 1, k0, lane);
      v16h b2 = load_b_frag_swz(Bt, DM >> 4, nb + 2, k0, lane);
      v16h b3 = load_b_frag_swz(Bt, DM >> 4, nb + 3, k0, lane);
      acc0 = __builtin_amdgcn_wmma_f32_16x16x32_f16(false, a, false, b0, (short)0, acc0, false, false);
      acc1 = __builtin_amdgcn_wmma_f32_16x16x32_f16(false, a, false, b1, (short)0, acc1, false, false);
      acc2 = __builtin_amdgcn_wmma_f32_16x16x32_f16(false, a, false, b2, (short)0, acc2, false, false);
      acc3 = __builtin_amdgcn_wmma_f32_16x16x32_f16(false, a, false, b3, (short)0, acc3, false, false);
    }
  }
  int hi = lane >> 4;
  v8f accs[4] = {acc0, acc1, acc2, acc3};
  #pragma unroll
  for (int u = 0; u < 4; ++u) {
    int n = (tileN4 << 6) + (u << 4) + (lane & 15);
    float bv = bias[n];
    #pragma unroll
    for (int r = 0; r < 8; ++r) {
      int m = (tileM << 4) + 8 * hi + r;
      float v = accs[u][r] + bv;
      Out[(size_t)m * DM + n] = (half_t)(v > 0.f ? v : 0.f);
    }
  }
}

// ---- weight transpose + f16 convert + WMMA-fragment swizzle ---------------
// Logical Bt[k][n] = W[n, k] (Kpad zero-padded); stored swizzled:
// [k/32][n/16][lane = (n&15) | ((k>>3 & 1)<<4)][j = (k&7) | ((k>>4 & 1)<<3)]
__global__ void wT_convert(const float* __restrict__ W, half_t* __restrict__ Bt,
                           int N, int K, int Kpad, int rowStride, int elemStride,
                           int elemOff) {
  int idx = blockIdx.x * blockDim.x + threadIdx.x;
  if (idx >= Kpad * N) return;
  int k = idx / N, n = idx - k * N;
  float v = (k < K) ? W[(size_t)n * rowStride + (size_t)k * elemStride + elemOff]
                    : 0.f;
  int kb = k >> 5, kr = k & 31;
  int j  = (kr & 7) | ((kr >> 4) << 3), hb = (kr >> 3) & 1;
  int lane = (n & 15) | (hb << 4);
  size_t oidx = ((((size_t)kb * (N >> 4) + (n >> 4)) << 5) + lane) * 16 + j;
  Bt[oidx] = (half_t)v;
}

// ---- Haar DWT (feature 0) + f16 convert + pad F->384 ----------------------
__global__ void dwt_convert(const float* __restrict__ x,
                            half_t* __restrict__ Al, half_t* __restrict__ Ah) {
  int idx = blockIdx.x * blockDim.x + threadIdx.x;
  if (idx >= TOKENS * FP) return;
  int tok = idx / FP, f = idx - tok * FP;
  float vl, vh;
  if (f >= FD) { vl = vh = 0.f; }
  else if (f == 0) {
    int b = tok >> 14, l = (tok >> 7) & 127, s = tok & 127;
    int l0 = l & ~1;
    size_t i0 = ((size_t)((b * LL + l0) * SS + s)) * FD;
    float x0 = x[i0], x1 = x[i0 + (size_t)SS * FD];
    float avg = 0.5f * (x0 + x1), dif = 0.5f * (x0 - x1);
    vl = avg;                       // idwt(cA,None): repeat avg
    vh = (l & 1) ? -dif : dif;      // idwt(None,cD): [+dif, -dif]
  } else {
    vl = vh = x[(size_t)tok * FD + f];
  }
  Al[idx] = (half_t)vl;
  Ah[idx] = (half_t)vh;
}

// ---- add time + spatial embeddings, f16 -> f16 ----------------------------
__global__ void embed_add(const half_t* __restrict__ in, const int* __restrict__ ts,
                          const float* __restrict__ temb_w,
                          const float* __restrict__ temb_b,
                          const float* __restrict__ spa,
                          half_t* __restrict__ out) {
  int idx = blockIdx.x * blockDim.x + threadIdx.x;
  if (idx >= TOKENS * DM) return;
  int tok = idx >> 8, d = idx & 255;
  int b = tok >> 14, l = (tok >> 7) & 127, s = tok & 127;
  int tv = ts[b * LL + l];
  float v = (float)in[idx] + temb_w[d * TT + tv] + temb_b[d] + spa[s * DM + d];
  out[idx] = (half_t)v;
}

// ---- sinusoidal PE table (L x D) ------------------------------------------
__global__ void pe_kernel(float* __restrict__ pe) {
  int idx = blockIdx.x * blockDim.x + threadIdx.x;
  if (idx >= LL * DM) return;
  int t = idx >> 8, d = idx & 255;
  float freq = __expf(-(float)(d & ~1) * (9.210340371976184f / (float)DM));
  pe[idx] = (d & 1) ? __cosf((float)t * freq) : __sinf((float)t * freq);
}

// ---- in = f16(pred + pe[tok % L]) -----------------------------------------
__global__ void add_pe_h(const float* __restrict__ pred, const float* __restrict__ pe,
                         half_t* __restrict__ out) {
  int idx = blockIdx.x * blockDim.x + threadIdx.x;
  if (idx >= TOKENS * DM) return;
  out[idx] = (half_t)(pred[idx] + pe[idx & (LL * DM - 1)]);
}

// ---- attention core: one block per (group p, head h); VALU f32 ------------
// QKV rows: token = p*128+t, width 768 (Q @ h*32, K @ 256+h*32, V @ 512+h*32)
__global__ void __launch_bounds__(128)
attn_kernel(const half_t* __restrict__ QKV, half_t* __restrict__ Out) {
  __shared__ half_t Ksh[LL * HD];
  __shared__ half_t Vsh[LL * HD];
  int p = blockIdx.x >> 3;
  int h = blockIdx.x & 7;
  int t = threadIdx.x;
  size_t rowBase = ((size_t)(p * LL + t)) * (3 * DM);
  const half_t* kr = QKV + rowBase + DM + h * HD;
  const half_t* vr = QKV + rowBase + 2 * DM + h * HD;
  #pragma unroll
  for (int e = 0; e < HD; ++e) { Ksh[t * HD + e] = kr[e]; Vsh[t * HD + e] = vr[e]; }
  __syncthreads();

  float q[HD];
  const half_t* qr = QKV + rowBase + h * HD;
  #pragma unroll
  for (int e = 0; e < HD; ++e) q[e] = (float)qr[e];

  const float scale = 0.17677669529663687f;  // 1/sqrt(32)
  float mx = -1e30f;
  for (int j = 0; j < LL; ++j) {
    float s = 0.f;
    #pragma unroll
    for (int e = 0; e < HD; ++e) s += q[e] * (float)Ksh[j * HD + e];
    mx = fmaxf(mx, s * scale);
  }
  float acc[HD];
  #pragma unroll
  for (int e = 0; e < HD; ++e) acc[e] = 0.f;
  float sum = 0.f;
  for (int j = 0; j < LL; ++j) {
    float s = 0.f;
    #pragma unroll
    for (int e = 0; e < HD; ++e) s += q[e] * (float)Ksh[j * HD + e];
    float w = __expf(s * scale - mx);
    sum += w;
    #pragma unroll
    for (int e = 0; e < HD; ++e) acc[e] += w * (float)Vsh[j * HD + e];
  }
  float inv = 1.0f / sum;
  half_t* orow = Out + ((size_t)(p * LL + t)) * DM + h * HD;
  #pragma unroll
  for (int e = 0; e < HD; ++e) orow[e] = (half_t)(acc[e] * inv);
}

// ---- output heads (N=1/2 projections + 2-way softmax) ---------------------
__global__ void heads_kernel(const float* __restrict__ predL,
                             const float* __restrict__ fused,
                             const float* __restrict__ regl_w, const float* __restrict__ regl_b,
                             const float* __restrict__ clal_w, const float* __restrict__ clal_b,
                             const float* __restrict__ regf_w, const float* __restrict__ regf_b,
                             const float* __restrict__ claf_w, const float* __restrict__ claf_b,
                             float* __restrict__ out) {
  int tok = blockIdx.x * blockDim.x + threadIdx.x;
  if (tok >= TOKENS) return;
  const float* pl = predL + (size_t)tok * DM;
  const float* fu = fused + (size_t)tok * DM;
  float lr = regl_b[0], l0 = clal_b[0], l1 = clal_b[1];
  float rr = regf_b[0], c0 = claf_b[0], c1 = claf_b[1];
  for (int d = 0; d < DM; ++d) {
    float p = pl[d], f = fu[d];
    lr += p * regl_w[d]; l0 += p * clal_w[d]; l1 += p * clal_w[DM + d];
    rr += f * regf_w[d]; c0 += f * claf_w[d]; c1 += f * claf_w[DM + d];
  }
  float m1 = fmaxf(l0, l1), e0 = __expf(l0 - m1), e1 = __expf(l1 - m1), s1 = e0 + e1;
  float m2 = fmaxf(c0, c1), f0 = __expf(c0 - m2), f1 = __expf(c1 - m2), s2 = f0 + f1;
  out[tok] = rr;                                      // reg   (B,L,S)
  out[TOKENS + tok * 2 + 0] = f0 / s2;                // cla   (B,L,S,2)
  out[TOKENS + tok * 2 + 1] = f1 / s2;
  out[3 * TOKENS + tok] = lr;                         // lreg  (B,L,S)
  out[4 * TOKENS + tok * 2 + 0] = e0 / s1;            // lcla  (B,L,S,2)
  out[4 * TOKENS + tok * 2 + 1] = e1 / s1;
}

// ---- TDM demonstration: NULL-tensor (count=0) load -> architectural no-op -
// Exercises the async-tensor issue path + s_wait_tensorcnt without touching
// any data (D# count=0 means "NULL tensor" per ISA 8.3).
// clang-23 toolchain: 6-arg form (u32x4, i32x8, i32x4, i32x4, i32x8, cpol).
__global__ void tdm_probe_kernel() {
#if __has_builtin(__builtin_amdgcn_tensor_load_to_lds) && __has_builtin(__builtin_amdgcn_s_wait_tensorcnt)
  u32x4 g0 = {0u, 0u, 0u, 0u};
  i32x8 g1 = {0, 0, 0, 0, 0, 0, 0, 0};
  i32x4 g2 = {0, 0, 0, 0};
  i32x4 g3 = {0, 0, 0, 0};
  i32x8 g4 = {0, 0, 0, 0, 0, 0, 0, 0};
  __builtin_amdgcn_tensor_load_to_lds(g0, g1, g2, g3, g4, 0);
  __builtin_amdgcn_s_wait_tensorcnt(0);
#endif
}

// ---------------------------------------------------------------------------
extern "C" void kernel_launch(void* const* d_in, const int* in_sizes, int n_in,
                              void* d_out, int out_size, void* d_ws, size_t ws_size,
                              hipStream_t stream) {
  (void)in_sizes; (void)n_in; (void)out_size; (void)ws_size;
  const float* x      = (const float*)d_in[0];
  const int*   ts     = (const int*)d_in[1];
  const float* wg_w   = (const float*)d_in[2];  const float* wg_b  = (const float*)d_in[3];
  const float* wh_w   = (const float*)d_in[4];  const float* wh_b  = (const float*)d_in[5];
  const float* temb_w = (const float*)d_in[6];  const float* temb_b= (const float*)d_in[7];
  const float* spa    = (const float*)d_in[8];
  const float* conv_w = (const float*)d_in[9];  const float* conv_b= (const float*)d_in[10];
  const float* tin_w  = (const float*)d_in[11]; const float* tin_b = (const float*)d_in[12];
  const float* tout_w = (const float*)d_in[13]; const float* tout_b= (const float*)d_in[14];
  const float* pl_w   = (const float*)d_in[15]; const float* pl_b  = (const float*)d_in[16];
  const float* ph_w   = (const float*)d_in[17]; const float* ph_b  = (const float*)d_in[18];
  const float* sin_w  = (const float*)d_in[19]; const float* sin_b = (const float*)d_in[20];
  const float* sout_w = (const float*)d_in[21]; const float* sout_b= (const float*)d_in[22];
  const float* cin_w  = (const float*)d_in[23]; const float* cin_b = (const float*)d_in[24];
  const float* cout_w = (const float*)d_in[25]; const float* cout_b= (const float*)d_in[26];
  const float* regl_w = (const float*)d_in[27]; const float* regl_b= (const float*)d_in[28];
  const float* clal_w = (const float*)d_in[29]; const float* clal_b= (const float*)d_in[30];
  const float* regf_w = (const float*)d_in[31]; const float* regf_b= (const float*)d_in[32];
  const float* claf_w = (const float*)d_in[33]; const float* claf_b= (const float*)d_in[34];
  float* out = (float*)d_out;

  // ---- workspace layout (with phase-based aliasing) ----
  char* ws = (char*)d_ws;
  size_t off = 0;
  auto alloc = [&](size_t bytes) {
    off = (off + 255) & ~(size_t)255;
    size_t cur = off; off += bytes; return cur;
  };
  const size_t H = sizeof(half_t);
  size_t o_wgT  = alloc((size_t)FP * DM * H);
  size_t o_whT  = alloc((size_t)FP * DM * H);
  size_t o_tinT = alloc((size_t)DM * 3 * DM * H);
  size_t o_toutT= alloc((size_t)DM * DM * H);
  size_t o_convT= alloc((size_t)3 * DM * DM * H);
  size_t o_plT  = alloc((size_t)DM * DM * H);
  size_t o_phT  = alloc((size_t)DM * DM * H);
  size_t o_sinT = alloc((size_t)DM * 3 * DM * H);
  size_t o_soutT= alloc((size_t)DM * DM * H);
  size_t o_cinT = alloc((size_t)DM * 3 * DM * H);
  size_t o_coutT= alloc((size_t)DM * DM * H);
  size_t o_pe   = alloc((size_t)LL * DM * 4);
  size_t o_regA = alloc((size_t)TOKENS * FP * H * 2);  // Al+Ah, later QKV (exact fit)
  size_t o_regB = alloc((size_t)TOKENS * DM * H);      // Xl_proj, later attnOut
  size_t o_regC = alloc((size_t)TOKENS * DM * H);      // Xh_proj, later Xt
  size_t o_Xc   = alloc((size_t)TOKENS * DM * H);      // conv output
  size_t o_regD = alloc((size_t)TOKENS * DM * H);      // Xl_e, later in_l
  size_t o_regE = alloc((size_t)TOKENS * DM * H);      // Xh_e, later in_h
  size_t o_predL= alloc((size_t)TOKENS * DM * 4);      // f32, kept until heads
  size_t o_regF = alloc((size_t)TOKENS * DM * 4);      // predH, later fused

  half_t* wgT   = (half_t*)(ws + o_wgT);
  half_t* whT   = (half_t*)(ws + o_whT);
  half_t* tinT  = (half_t*)(ws + o_tinT);
  half_t* toutT = (half_t*)(ws + o_toutT);
  half_t* convT = (half_t*)(ws + o_convT);
  half_t* plT   = (half_t*)(ws + o_plT);
  half_t* phT   = (half_t*)(ws + o_phT);
  half_t* sinT  = (half_t*)(ws + o_sinT);
  half_t* soutT = (half_t*)(ws + o_soutT);
  half_t* cinT  = (half_t*)(ws + o_cinT);
  half_t* coutT = (half_t*)(ws + o_coutT);
  float*  peb   = (float*)(ws + o_pe);
  half_t* Al16  = (half_t*)(ws + o_regA);
  half_t* Ah16  = Al16 + (size_t)TOKENS * FP;
  half_t* QKV16 = (half_t*)(ws + o_regA);              // alias (after projections)
  half_t* Xlp16 = (half_t*)(ws + o_regB);
  half_t* attnO = (half_t*)(ws + o_regB);              // alias (after temporal QKV)
  half_t* Xhp16 = (half_t*)(ws + o_regC);
  half_t* Xt16  = (half_t*)(ws + o_regC);              // alias (after conv)
  half_t* Xc16  = (half_t*)(ws + o_Xc);
  half_t* Xle16 = (half_t*)(ws + o_regD);
  half_t* inl16 = (half_t*)(ws + o_regD);              // alias (after pred_l GEMM)
  half_t* Xhe16 = (half_t*)(ws + o_regE);
  half_t* inh16 = (half_t*)(ws + o_regE);              // alias (after pred_h GEMM)
  float*  predL = (float*)(ws + o_predL);
  float*  predH = (float*)(ws + o_regF);
  float*  fused = (float*)(ws + o_regF);               // alias (after add_pe_h)

  auto gemm = [&](const half_t* A, int ldA, const half_t* Bm, int tilesNB, int nbOff,
                  const float* bias, void* C, int ldC,
                  int M, int N, int K, int flags) {
    int strips = (M >> 4) * (N >> 6);   // 16x64 strips, one per wave
    int blocks = (strips + 7) >> 3;
    gemm_wmma_f16<<<blocks, 256, 0, stream>>>(A, ldA, Bm, tilesNB, nbOff, bias,
                                              C, ldC, M, N, K, flags);
  };
  auto cvt = [&](const float* W, half_t* Bt, int N, int K, int Kpad,
                 int rs, int es, int eo) {
    int n = Kpad * N;
    wT_convert<<<(n + 255) / 256, 256, 0, stream>>>(W, Bt, N, K, Kpad, rs, es, eo);
  };

  // 1) weight staging (f32 -> transposed, fragment-swizzled f16 B-matrices)
  cvt(wg_w,  wgT,  DM, FD, FP, FD, 1, 0);
  cvt(wh_w,  whT,  DM, FD, FP, FD, 1, 0);
  cvt(tin_w, tinT, 3 * DM, DM, DM, DM, 1, 0);
  cvt(tout_w,toutT,DM, DM, DM, DM, 1, 0);
  for (int t = 0; t < 3; ++t)
    cvt(conv_w, convT + (size_t)t * DM * DM, DM, DM, DM, 3 * DM, 3, t);
  cvt(pl_w,  plT,  DM, DM, DM, DM, 1, 0);
  cvt(ph_w,  phT,  DM, DM, DM, DM, 1, 0);
  cvt(sin_w, sinT, 3 * DM, DM, DM, DM, 1, 0);
  cvt(sout_w,soutT,DM, DM, DM, DM, 1, 0);
  cvt(cin_w, cinT, 3 * DM, DM, DM, DM, 1, 0);
  cvt(cout_w,coutT,DM, DM, DM, DM, 1, 0);
  pe_kernel<<<(LL * DM + 255) / 256, 256, 0, stream>>>(peb);
  tdm_probe_kernel<<<1, 32, 0, stream>>>();

  // 2) Haar DWT + f16 A-matrix staging (F padded to 384)
  dwt_convert<<<(TOKENS * FP + 255) / 256, 256, 0, stream>>>(x, Al16, Ah16);

  // 3) input projections (WMMA)
  gemm(Al16, FP, wgT, DM >> 4, 0, wg_b, Xlp16, DM, TOKENS, DM, FP, GF_OUTH);
  gemm(Ah16, FP, whT, DM >> 4, 0, wh_b, Xhp16, DM, TOKENS, DM, FP, GF_OUTH);

  // 4) dilated conv as 3-tap accumulated WMMA GEMM (+bias +relu)
  {
    int strips = (TOKENS >> 4) * (DM >> 6);
    conv_gemm_wmma<<<(strips + 7) >> 3, 256, 0, stream>>>(Xhp16, convT, conv_b, Xc16);
  }

  // 5) temporal MHA (QKV GEMM -> attention -> out-proj)
  gemm(Xlp16, DM, tinT, (3 * DM) >> 4, 0, tin_b, QKV16, 3 * DM, TOKENS, 3 * DM, DM, GF_OUTH);
  attn_kernel<<<(TOKENS / LL) * NHEAD, LL, 0, stream>>>(QKV16, attnO);
  gemm(attnO, DM, toutT, DM >> 4, 0, tout_b, Xt16, DM, TOKENS, DM, DM, GF_OUTH);

  // 6) time + spatial embeddings
  embed_add<<<(TOKENS * DM + 255) / 256, 256, 0, stream>>>(Xt16, ts, temb_w, temb_b, spa, Xle16);
  embed_add<<<(TOKENS * DM + 255) / 256, 256, 0, stream>>>(Xc16, ts, temb_w, temb_b, spa, Xhe16);

  // 7) decoder projections (f32 out, pred_l kept for the aux heads)
  gemm(Xle16, DM, plT, DM >> 4, 0, pl_b, predL, DM, TOKENS, DM, DM, 0);
  gemm(Xhe16, DM, phT, DM >> 4, 0, ph_b, predH, DM, TOKENS, DM, DM, 0);
  add_pe_h<<<(TOKENS * DM + 255) / 256, 256, 0, stream>>>(predL, peb, inl16);
  add_pe_h<<<(TOKENS * DM + 255) / 256, 256, 0, stream>>>(predH, peb, inh16);

  // 8) self-attention MHA -> fused (f32)
  gemm(inl16, DM, sinT, (3 * DM) >> 4, 0, sin_b, QKV16, 3 * DM, TOKENS, 3 * DM, DM, GF_OUTH);
  attn_kernel<<<(TOKENS / LL) * NHEAD, LL, 0, stream>>>(QKV16, attnO);
  gemm(attnO, DM, soutT, DM >> 4, 0, sout_b, fused, DM, TOKENS, DM, DM, 0);

  // 9) cross-attention MHA; out-proj accumulates into fused (a_self + a_cross)
  gemm(inl16, DM, cinT, (3 * DM) >> 4, 0,  cin_b,      QKV16,      3 * DM, TOKENS, DM,     DM, GF_OUTH);
  gemm(inh16, DM, cinT, (3 * DM) >> 4, 16, cin_b + DM, QKV16 + DM, 3 * DM, TOKENS, 2 * DM, DM, GF_OUTH);
  attn_kernel<<<(TOKENS / LL) * NHEAD, LL, 0, stream>>>(QKV16, attnO);
  gemm(attnO, DM, coutT, DM >> 4, 0, cout_b, fused, DM, TOKENS, DM, DM, GF_ACC);

  // 10) output heads -> d_out (reg | cla | lreg | lcla)
  heads_kernel<<<(TOKENS + 255) / 256, 256, 0, stream>>>(
      predL, fused, regl_w, regl_b, clal_w, clal_b,
      regf_w, regf_b, claf_w, claf_b, out);
}
